// GraphLayer_51642686767329
// MI455X (gfx1250) — matrix operand order repaired and checked by reading, not compiled, not run on hardware
//
#include <hip/hip_runtime.h>
#include <hip/hip_bf16.h>
#include <stdint.h>

// ---------------------------------------------------------------------------
// CDNA5 (gfx1250) bf16 WMMA GEMM pipeline for the GCN-GRU reference.
// All matmuls run on v_wmma_f32_16x16x32_bf16 (wave32), f32 accumulate.
// B-operands are stored K-major so every fragment load is contiguous b128s.
// support (256 MB, streamed once per step) is loaded non-temporally and
// converted f32->bf16 in registers with packed converts.
// ---------------------------------------------------------------------------

typedef __attribute__((ext_vector_type(16))) __bf16 v16bf;
typedef __attribute__((ext_vector_type(8)))  float  v8f;
typedef __attribute__((ext_vector_type(4)))  float  v4f;

union FragBF {
    v16bf v;
    uint4 q[2];
    unsigned int u[8];
    unsigned short s[16];
};

__device__ __forceinline__ unsigned short f32_to_bf16_rne(float f) {
    union { float f; unsigned int u; } c; c.f = f;
    unsigned int u = c.u;
    unsigned int r = (u + 0x7FFFu + ((u >> 16) & 1u)) >> 16;
    return (unsigned short)r;
}

// Pack two f32 into two bf16 in one DWORD ({lo=a, hi=b}).
__device__ __forceinline__ unsigned int pack2_bf16(float a, float b) {
#if defined(__has_builtin) && __has_builtin(__builtin_amdgcn_cvt_pk_bf16_f32)
    typedef __attribute__((ext_vector_type(2))) __bf16 v2bf;
    union { v2bf v; unsigned int u; } c;
    c.v = __builtin_amdgcn_cvt_pk_bf16_f32(a, b);
    return c.u;
#else
    union { float f; unsigned int u; } ca, cb;
    ca.f = a; cb.f = b;
    unsigned int ua = ca.u + 0x7FFFu + ((ca.u >> 16) & 1u);
    unsigned int ub = cb.u + 0x7FFFu + ((cb.u >> 16) & 1u);
    // v_perm_b32: take high halves of {ub (hi), ua (lo)} -> {ub.hi, ua.hi}
    return __builtin_amdgcn_perm(ub, ua, 0x07060302u);
#endif
}

// A-fragment (16x32 bf16, MxK) from a row-major f32 row, NT loads + packed cvt.
// Lane group g=lane>>4: elems 0..7 <- K = k0+8g+e ; elems 8..15 <- K = k0+16+8g+e
__device__ __forceinline__ v16bf load_afrag_f32_nt(const float* __restrict__ Arow, int k0, int g) {
    const v4f* p0 = (const v4f*)(Arow + k0 + 8 * g);
    const v4f* p1 = (const v4f*)(Arow + k0 + 16 + 8 * g);
    v4f f0 = __builtin_nontemporal_load(p0);
    v4f f1 = __builtin_nontemporal_load(p0 + 1);
    v4f f2 = __builtin_nontemporal_load(p1);
    v4f f3 = __builtin_nontemporal_load(p1 + 1);
    FragBF f;
    f.u[0] = pack2_bf16(f0[0], f0[1]);
    f.u[1] = pack2_bf16(f0[2], f0[3]);
    f.u[2] = pack2_bf16(f1[0], f1[1]);
    f.u[3] = pack2_bf16(f1[2], f1[3]);
    f.u[4] = pack2_bf16(f2[0], f2[1]);
    f.u[5] = pack2_bf16(f2[2], f2[3]);
    f.u[6] = pack2_bf16(f3[0], f3[1]);
    f.u[7] = pack2_bf16(f3[2], f3[3]);
    return f.v;
}

// A-fragment from a row-major bf16 row: two contiguous 16B loads.
__device__ __forceinline__ v16bf load_afrag_bf16(const unsigned short* __restrict__ Arow, int k0, int g) {
    FragBF f;
    f.q[0] = *(const uint4*)(Arow + k0 + 8 * g);
    f.q[1] = *(const uint4*)(Arow + k0 + 16 + 8 * g);
    return f.v;
}

// B-fragment (32x16 bf16, KxN) from a K-major (transposed) bf16 row:
// lane (col) reads K = k0+8g .. k0+8g+7 and k0+16+8g .. k0+16+8g+7 (two 16B loads).
__device__ __forceinline__ v16bf load_bfrag(const unsigned short* __restrict__ BTrow, int k0, int g) {
    FragBF f;
    f.q[0] = *(const uint4*)(BTrow + k0 + 8 * g);
    f.q[1] = *(const uint4*)(BTrow + k0 + 16 + 8 * g);
    return f.v;
}

#define WMMA_BF16(A, B, C) \
    __builtin_amdgcn_wmma_f32_16x16x32_bf16(false, (A), false, (B), (short)0, (C), false, false)

__device__ __forceinline__ v8f zero_v8f() {
    v8f z = {0.f, 0.f, 0.f, 0.f, 0.f, 0.f, 0.f, 0.f};
    return z;
}

static constexpr int Bb = 16, Nn = 2048, Dd = 128;

// ---------------------------------------------------------------------------
// Weight convert: f32 [128,128] row-major -> bf16 [128,128] K-major (WT[o][i])
// ---------------------------------------------------------------------------
__global__ __launch_bounds__(256) void k_wcvt(const float* __restrict__ W,
                                              unsigned short* __restrict__ WT) {
    int t = blockIdx.x * 256 + threadIdx.x;   // 64 blocks * 256 = 16384
    int i = t >> 7, o = t & 127;
    WT[o * 128 + i] = f32_to_bf16_rne(W[i * 128 + o]);
}

// ---------------------------------------------------------------------------
// Encode: h = relu(x @ Wenc + benc). Writes h f32, h bf16 (row-major), hT bf16.
// Block = 8 waves, each wave a 16x16 tile; grid = (32768/128)*8.
// ---------------------------------------------------------------------------
__global__ __launch_bounds__(256) void k_encode(const float* __restrict__ x,
                                                const unsigned short* __restrict__ WT,
                                                const float* __restrict__ bias,
                                                float* __restrict__ hF,
                                                unsigned short* __restrict__ hB,
                                                unsigned short* __restrict__ hT) {
    const int lane = threadIdx.x & 31, w = threadIdx.x >> 5;
    const int g = lane >> 4, lr = lane & 15;
    const int cs = blockIdx.x & 7, rb = blockIdx.x >> 3;
    const int mBase = rb * 128 + w * 16;
    const int col = cs * 16 + lr;

    const float* Arow = x + (size_t)(mBase + lr) * Dd;
    const unsigned short* Brow = WT + (size_t)col * Dd;

    v8f acc = zero_v8f();
#pragma unroll
    for (int k0 = 0; k0 < Dd; k0 += 32) {
        acc = WMMA_BF16(load_afrag_f32_nt(Arow, k0, g), load_bfrag(Brow, k0, g), acc);
    }
    const float bb = bias[col];
#pragma unroll
    for (int i = 0; i < 8; ++i) {
        int m = mBase + 8 * g + i;
        float v = acc[i] + bb;
        v = v > 0.f ? v : 0.f;
        size_t idx = (size_t)m * Dd + col;
        unsigned short bv = f32_to_bf16_rne(v);
        hF[idx] = v;
        hB[idx] = bv;
        int bi = m >> 11, n = m & (Nn - 1);
        hT[((size_t)bi * Dd + col) * Nn + n] = bv;
    }
}

// ---------------------------------------------------------------------------
// Aggregate: a = (support @ h) * (1 + ffd), written as bf16 row-major.
// Each wave owns a 16-row x 128-col stripe (8 accumulators) so every
// support element is streamed from HBM exactly once (NT loads).
// Grid = B*16 = 256 blocks of 8 waves.
// ---------------------------------------------------------------------------
__global__ __launch_bounds__(256) void k_aggregate(const float* __restrict__ support,
                                                   const unsigned short* __restrict__ hT,
                                                   const float* __restrict__ ffd,
                                                   unsigned short* __restrict__ aB) {
    const int lane = threadIdx.x & 31, w = threadIdx.x >> 5;
    const int g = lane >> 4, lr = lane & 15;
    const int b = blockIdx.x >> 4;
    const int nBase = (blockIdx.x & 15) * 128 + w * 16;

    const float* Arow = support + ((size_t)b * Nn + nBase + lr) * Nn;
    const unsigned short* hTb = hT + (size_t)b * Dd * Nn;
    const unsigned short* Brow[8];
#pragma unroll
    for (int s = 0; s < 8; ++s) Brow[s] = hTb + (size_t)(s * 16 + lr) * Nn;

    v8f acc[8];
#pragma unroll
    for (int s = 0; s < 8; ++s) acc[s] = zero_v8f();

#pragma unroll 4
    for (int k0 = 0; k0 < Nn; k0 += 32) {
        v16bf a = load_afrag_f32_nt(Arow, k0, g);
#pragma unroll
        for (int s = 0; s < 8; ++s) {
            acc[s] = WMMA_BF16(a, load_bfrag(Brow[s], k0, g), acc[s]);
        }
    }
#pragma unroll
    for (int i = 0; i < 8; ++i) {
        int n = nBase + 8 * g + i;
        float scale = 1.0f + ffd[(size_t)b * Nn + n];
        unsigned short* dst = aB + ((size_t)b * Nn + n) * Dd + lr;
#pragma unroll
        for (int s = 0; s < 8; ++s) dst[s * 16] = f32_to_bf16_rne(acc[s][i] * scale);
    }
}

// ---------------------------------------------------------------------------
// Gates z, r:  z = sig(a@Wz0 + out@Wz1 + bz),  r = sig(a@Wr0 + out@Wr1 + br)
// Writes z f32 and rh = bf16(r * out) row-major.
// ---------------------------------------------------------------------------
__global__ __launch_bounds__(256) void k_zr(const unsigned short* __restrict__ aB,
                                            const unsigned short* __restrict__ hB,
                                            const unsigned short* __restrict__ Wz0T,
                                            const unsigned short* __restrict__ Wz1T,
                                            const unsigned short* __restrict__ Wr0T,
                                            const unsigned short* __restrict__ Wr1T,
                                            const float* __restrict__ bz0,
                                            const float* __restrict__ bz1,
                                            const float* __restrict__ br0,
                                            const float* __restrict__ br1,
                                            const float* __restrict__ hF,
                                            float* __restrict__ zF,
                                            unsigned short* __restrict__ rhB) {
    const int lane = threadIdx.x & 31, w = threadIdx.x >> 5;
    const int g = lane >> 4, lr = lane & 15;
    const int cs = blockIdx.x & 7, rb = blockIdx.x >> 3;
    const int mBase = rb * 128 + w * 16;
    const int col = cs * 16 + lr;

    const unsigned short* Aa = aB + (size_t)(mBase + lr) * Dd;
    const unsigned short* Ah = hB + (size_t)(mBase + lr) * Dd;
    const unsigned short* Bz0 = Wz0T + (size_t)col * Dd;
    const unsigned short* Bz1 = Wz1T + (size_t)col * Dd;
    const unsigned short* Br0 = Wr0T + (size_t)col * Dd;
    const unsigned short* Br1 = Wr1T + (size_t)col * Dd;

    v8f accZ = zero_v8f(), accR = zero_v8f();
#pragma unroll
    for (int k0 = 0; k0 < Dd; k0 += 32) {
        v16bf af = load_afrag_bf16(Aa, k0, g);
        v16bf hf = load_afrag_bf16(Ah, k0, g);
        accZ = WMMA_BF16(af, load_bfrag(Bz0, k0, g), accZ);
        accR = WMMA_BF16(af, load_bfrag(Br0, k0, g), accR);
        accZ = WMMA_BF16(hf, load_bfrag(Bz1, k0, g), accZ);
        accR = WMMA_BF16(hf, load_bfrag(Br1, k0, g), accR);
    }
    const float zb = bz0[col] + bz1[col];
    const float rb_ = br0[col] + br1[col];
#pragma unroll
    for (int i = 0; i < 8; ++i) {
        size_t idx = (size_t)(mBase + 8 * g + i) * Dd + col;
        float z = 1.f / (1.f + __expf(-(accZ[i] + zb)));
        float r = 1.f / (1.f + __expf(-(accR[i] + rb_)));
        float out = hF[idx];
        zF[idx] = z;
        rhB[idx] = f32_to_bf16_rne(r * out);
    }
}

// ---------------------------------------------------------------------------
// Candidate + update: hh = relu(a@Wh0 + rh@Wh1 + bh); hnew = hh*z + out*(1-z)
// Updates hF/hB/hT state; optionally mirrors to outF (final step -> d_out).
// ---------------------------------------------------------------------------
__global__ __launch_bounds__(256) void k_hnew(const unsigned short* __restrict__ aB,
                                              const unsigned short* __restrict__ rhB,
                                              const unsigned short* __restrict__ Wh0T,
                                              const unsigned short* __restrict__ Wh1T,
                                              const float* __restrict__ bh0,
                                              const float* __restrict__ bh1,
                                              const float* __restrict__ zF,
                                              float* __restrict__ hF,
                                              unsigned short* __restrict__ hB,
                                              unsigned short* __restrict__ hT,
                                              float* __restrict__ outF) {
    const int lane = threadIdx.x & 31, w = threadIdx.x >> 5;
    const int g = lane >> 4, lr = lane & 15;
    const int cs = blockIdx.x & 7, rb = blockIdx.x >> 3;
    const int mBase = rb * 128 + w * 16;
    const int col = cs * 16 + lr;

    const unsigned short* Aa = aB + (size_t)(mBase + lr) * Dd;
    const unsigned short* Ar = rhB + (size_t)(mBase + lr) * Dd;
    const unsigned short* B0 = Wh0T + (size_t)col * Dd;
    const unsigned short* B1 = Wh1T + (size_t)col * Dd;

    v8f acc = zero_v8f();
#pragma unroll
    for (int k0 = 0; k0 < Dd; k0 += 32) {
        acc = WMMA_BF16(load_afrag_bf16(Aa, k0, g), load_bfrag(B0, k0, g), acc);
        acc = WMMA_BF16(load_afrag_bf16(Ar, k0, g), load_bfrag(B1, k0, g), acc);
    }
    const float hb = bh0[col] + bh1[col];
#pragma unroll
    for (int i = 0; i < 8; ++i) {
        int m = mBase + 8 * g + i;
        size_t idx = (size_t)m * Dd + col;
        float hh = acc[i] + hb;
        hh = hh > 0.f ? hh : 0.f;
        float z = zF[idx];
        float out = hF[idx];
        float hn = hh * z + out * (1.f - z);
        unsigned short bv = f32_to_bf16_rne(hn);
        hF[idx] = hn;
        hB[idx] = bv;
        int bi = m >> 11, n = m & (Nn - 1);
        hT[((size_t)bi * Dd + col) * Nn + n] = bv;
        if (outF) outF[idx] = hn;
    }
}

// ---------------------------------------------------------------------------
extern "C" void kernel_launch(void* const* d_in, const int* in_sizes, int n_in,
                              void* d_out, int out_size, void* d_ws, size_t ws_size,
                              hipStream_t stream) {
    (void)in_sizes; (void)n_in; (void)out_size; (void)ws_size;

    const float* x    = (const float*)d_in[0];
    const float* sup  = (const float*)d_in[1];
    const float* ffd  = (const float*)d_in[2];
    /* d_in[3] = mask, unused (all-ones / overwritten in source) */
    const float* Wenc = (const float*)d_in[4];
    const float* benc = (const float*)d_in[5];
    const float* Wz0  = (const float*)d_in[6];
    const float* bz0  = (const float*)d_in[7];
    const float* Wz1  = (const float*)d_in[8];
    const float* bz1  = (const float*)d_in[9];
    const float* Wr0  = (const float*)d_in[10];
    const float* br0  = (const float*)d_in[11];
    const float* Wr1  = (const float*)d_in[12];
    const float* br1  = (const float*)d_in[13];
    const float* Wh0  = (const float*)d_in[14];
    const float* bh0  = (const float*)d_in[15];
    const float* Wh1  = (const float*)d_in[16];
    const float* bh1  = (const float*)d_in[17];

    const size_t M = (size_t)Bb * Nn;   // 32768 rows

    uintptr_t p = (uintptr_t)d_ws;
    auto take = [&](size_t bytes) -> void* {
        p = (p + 255) & ~(uintptr_t)255;
        void* r = (void*)p;
        p += bytes;
        return r;
    };
    float*          hF  = (float*)take(M * Dd * sizeof(float));           // 16 MB
    float*          zF  = (float*)take(M * Dd * sizeof(float));           // 16 MB
    unsigned short* hB  = (unsigned short*)take(M * Dd * 2);              //  8 MB
    unsigned short* hT  = (unsigned short*)take((size_t)Bb * Dd * Nn * 2);//  8 MB
    unsigned short* aB  = (unsigned short*)take(M * Dd * 2);              //  8 MB
    unsigned short* rhB = (unsigned short*)take(M * Dd * 2);              //  8 MB
    unsigned short* WencT = (unsigned short*)take(Dd * Dd * 2);
    unsigned short* Wz0T  = (unsigned short*)take(Dd * Dd * 2);
    unsigned short* Wz1T  = (unsigned short*)take(Dd * Dd * 2);
    unsigned short* Wr0T  = (unsigned short*)take(Dd * Dd * 2);
    unsigned short* Wr1T  = (unsigned short*)take(Dd * Dd * 2);
    unsigned short* Wh0T  = (unsigned short*)take(Dd * Dd * 2);
    unsigned short* Wh1T  = (unsigned short*)take(Dd * Dd * 2);

    // Weight conversion to bf16, K-major.
    k_wcvt<<<64, 256, 0, stream>>>(Wenc, WencT);
    k_wcvt<<<64, 256, 0, stream>>>(Wz0, Wz0T);
    k_wcvt<<<64, 256, 0, stream>>>(Wz1, Wz1T);
    k_wcvt<<<64, 256, 0, stream>>>(Wr0, Wr0T);
    k_wcvt<<<64, 256, 0, stream>>>(Wr1, Wr1T);
    k_wcvt<<<64, 256, 0, stream>>>(Wh0, Wh0T);
    k_wcvt<<<64, 256, 0, stream>>>(Wh1, Wh1T);

    const int gSmall = (int)(M / 128) * 8;   // 2048 blocks
    k_encode<<<gSmall, 256, 0, stream>>>(x, WencT, benc, hF, hB, hT);

    for (int step = 0; step < 2; ++step) {
        k_aggregate<<<Bb * 16, 256, 0, stream>>>(sup, hT, ffd, aB);
        k_zr<<<gSmall, 256, 0, stream>>>(aB, hB, Wz0T, Wz1T, Wr0T, Wr1T,
                                         bz0, bz1, br0, br1, hF, zF, rhB);
        k_hnew<<<gSmall, 256, 0, stream>>>(aB, rhB, Wh0T, Wh1T, bh0, bh1, zF,
                                           hF, hB, hT,
                                           step == 1 ? (float*)d_out : nullptr);
    }
}
